// NeuralODEPKPD_16080357556486
// MI455X (gfx1250) — matrix operand order repaired and checked by reading, not compiled, not run on hardware
//
#include <hip/hip_runtime.h>

typedef __attribute__((ext_vector_type(16))) _Float16 v16h;
typedef __attribute__((ext_vector_type(8)))  _Float16 v8h;
typedef __attribute__((ext_vector_type(8)))  float    v8f;

constexpr int kHid   = 256;
constexpr int kInDim = 64;
constexpr int kOut   = 4;
constexpr int kB     = 2048;
constexpr int kT     = 128;
constexpr int kRows  = 16;    // batch rows per workgroup
constexpr int kThreads = 256; // 8 waves: 2 per SIMD32 for latency hiding
constexpr int kTilesPerWave = 2; // 8 waves x 2 N-tiles = 16 N-tiles (N=256)

// LDS layout (bytes)
constexpr int OFF_WT1 = 0;
constexpr int OFF_WT2 = kHid * kHid * 2;
constexpr int OFF_Y   = 2 * kHid * kHid * 2;
constexpr int OFF_T1  = OFF_Y  + kRows * kHid * 2;
constexpr int OFF_HF  = OFF_T1 + kRows * kHid * 2;
constexpr int SMEM_BYTES = OFF_HF + kRows * kHid * 4;   // 294912 B <= 320KB WGP LDS

// One 16x256 (MxK) x 256x256 tile-GEMM for the 16 rows owned by this
// workgroup; each wave produces 2 N-tiles (32 columns). f16 in, f32 accum.
__device__ __forceinline__ void gemm16(const _Float16* __restrict__ src,  // LDS [16][256] f16 (A)
                                       const _Float16* __restrict__ Wt,   // LDS [256 n][256 k] f16 (B^T)
                                       const float* bn,                   // per-lane bias b[n_i]
                                       v8f acc[kTilesPerWave],
                                       int lo, int hi, int wave)
{
#pragma unroll
    for (int i = 0; i < kTilesPerWave; ++i) {
        v8f c;
#pragma unroll
        for (int r = 0; r < 8; ++r) c[r] = bn[i];
        acc[i] = c;
    }
#pragma unroll
    for (int kc = 0; kc < 8; ++kc) {
        // A fragment (16x32 f16): lane lo holds row M=lo;
        // element e -> K = kc*32 + hi*16 + e
        const _Float16* ab = src + lo * kHid + kc * 32 + hi * 16;
        v8h a0 = *(const v8h*)(ab);
        v8h a1 = *(const v8h*)(ab + 8);
        v16h a;
#pragma unroll
        for (int e = 0; e < 8; ++e) { a[e] = a0[e]; a[8 + e] = a1[e]; }
#pragma unroll
        for (int i = 0; i < kTilesPerWave; ++i) {
            // B fragment (32x16 f16): lane lo holds column N; element e -> K = kc*32 + hi*16 + e
            int n = (wave * kTilesPerWave + i) * 16 + lo;
            v16h b = *(const v16h*)(Wt + n * kHid + kc * 32 + hi * 16);
            acc[i] = __builtin_amdgcn_wmma_f32_16x16x32_f16(
                false, a, false, b, (short)0, acc[i], false, false);
        }
    }
}

extern "C" __global__ __launch_bounds__(kThreads, 1)
void node_rk4_kernel(const float* __restrict__ x,      // [B, 64]
                     const float* __restrict__ times,  // [T]
                     const float* __restrict__ W_in, const float* __restrict__ b_in,
                     const float* __restrict__ W1,   const float* __restrict__ b1,
                     const float* __restrict__ W2,   const float* __restrict__ b2,
                     const float* __restrict__ W_pk, const float* __restrict__ b_pk,
                     const float* __restrict__ W_pd, const float* __restrict__ b_pd,
                     float* __restrict__ out)
{
    extern __shared__ char smem[];
    _Float16* Wt1 = (_Float16*)(smem + OFF_WT1);
    _Float16* Wt2 = (_Float16*)(smem + OFF_WT2);
    _Float16* y   = (_Float16*)(smem + OFF_Y);    // current RK4 stage input (f16)
    _Float16* t1f = (_Float16*)(smem + OFF_T1);   // tanh intermediate (f16)
    float*    hf  = (float*)(smem + OFF_HF);      // full-precision h for projections

    const int tid  = threadIdx.x;
    const int wave = tid >> 5;
    const int lane = tid & 31;
    const int lo   = lane & 15;
    const int hi   = lane >> 4;
    const int rowbase = blockIdx.x * kRows;

    // Stage weights into LDS once, transposed (Wt[n][k] = W[k][n]) as f16.
    for (int idx = tid; idx < kHid * kHid; idx += kThreads) {
        int k = idx >> 8, n = idx & 255;
        Wt1[n * kHid + k] = (_Float16)W1[idx];
        Wt2[n * kHid + k] = (_Float16)W2[idx];
    }

    // This lane's owned columns (fixed for the whole kernel) + biases.
    int   ncol[kTilesPerWave];
    float b1v[kTilesPerWave], b2v[kTilesPerWave];
#pragma unroll
    for (int i = 0; i < kTilesPerWave; ++i) {
        ncol[i] = (wave * kTilesPerWave + i) * 16 + lo;
        b1v[i]  = b1[ncol[i]];
        b2v[i]  = b2[ncol[i]];
    }

    // h0 = x @ W_in + b_in (fp32 VALU, once). Ownership matches WMMA C/D layout:
    // lane element (i, r) <-> (m = r + hi*8, n = ncol[i]).
    float hreg[kTilesPerWave * 8];
#pragma unroll
    for (int i = 0; i < kTilesPerWave; ++i) {
        int n = ncol[i];
#pragma unroll
        for (int r = 0; r < 8; ++r) {
            int m = r + hi * 8;
            float s = b_in[n];
            for (int k = 0; k < kInDim; ++k)
                s += x[(rowbase + m) * kInDim + k] * W_in[k * kHid + n];
            hreg[i * 8 + r] = s;
        }
    }

    // Publish h -> y (f16 WMMA operand) and hf (f32 for projections).
#pragma unroll
    for (int i = 0; i < kTilesPerWave; ++i)
#pragma unroll
        for (int r = 0; r < 8; ++r) {
            int m = r + hi * 8, n = ncol[i];
            y[m * kHid + n]  = (_Float16)hreg[i * 8 + r];
            hf[m * kHid + n] = hreg[i * 8 + r];
        }
    __syncthreads();   // also covers Wt1/Wt2 staging

    // pk_pred at t=0
    if (tid < 64) {
        int m = tid >> 2, j = tid & 3;
        float s = b_pk[j];
        for (int k = 0; k < kHid; ++k) s += hf[m * kHid + k] * W_pk[k * kOut + j];
        out[((size_t)(rowbase + m) * kT + 0) * kOut + j] = s;
    }

#pragma unroll 1
    for (int t = 1; t < kT; ++t) {
        float dt = times[t] - times[t - 1];
        float accrk[kTilesPerWave * 8];
#pragma unroll
        for (int q = 0; q < kTilesPerWave * 8; ++q) accrk[q] = 0.f;

#pragma unroll 1
        for (int s = 0; s < 4; ++s) {
            // layer 1: t1 = tanh(y @ W1 + b1)
            v8f a1v[kTilesPerWave];
            gemm16(y, Wt1, b1v, a1v, lo, hi, wave);
#pragma unroll
            for (int i = 0; i < kTilesPerWave; ++i)
#pragma unroll
                for (int r = 0; r < 8; ++r) {
                    int m = r + hi * 8, n = ncol[i];
                    t1f[m * kHid + n] = (_Float16)tanhf(a1v[i][r]);
                }
            __syncthreads();

            // layer 2: k = t1 @ W2 + b2 ; RK4 combine (register-local)
            v8f a2v[kTilesPerWave];
            gemm16(t1f, Wt2, b2v, a2v, lo, hi, wave);
            float wgt = (s == 1 || s == 2) ? 2.f : 1.f;
            float cdt = (s < 2) ? 0.5f * dt : ((s == 2) ? dt : 0.f);
#pragma unroll
            for (int i = 0; i < kTilesPerWave; ++i)
#pragma unroll
                for (int r = 0; r < 8; ++r) {
                    int q = i * 8 + r;
                    float kv = a2v[i][r];
                    accrk[q] += wgt * kv;
                    if (s < 3) {
                        int m = r + hi * 8, n = ncol[i];
                        y[m * kHid + n] = (_Float16)(hreg[q] + cdt * kv);
                    }
                }
            __syncthreads();
        }

        // h <- h + dt/6 * (k1 + 2k2 + 2k3 + k4); republish
#pragma unroll
        for (int i = 0; i < kTilesPerWave; ++i)
#pragma unroll
            for (int r = 0; r < 8; ++r) {
                int q = i * 8 + r;
                hreg[q] += (dt * (1.f / 6.f)) * accrk[q];
                int m = r + hi * 8, n = ncol[i];
                y[m * kHid + n]  = (_Float16)hreg[q];
                hf[m * kHid + n] = hreg[q];
            }
        __syncthreads();

        // pk_pred at step t
        if (tid < 64) {
            int m = tid >> 2, j = tid & 3;
            float ssum = b_pk[j];
            for (int k = 0; k < kHid; ++k) ssum += hf[m * kHid + k] * W_pk[k * kOut + j];
            out[((size_t)(rowbase + m) * kT + t) * kOut + j] = ssum;
        }
    }

    // pd_pred from final h
    if (tid < 16) {
        float ssum = b_pd[0];
        for (int k = 0; k < kHid; ++k) ssum += hf[tid * kHid + k] * W_pd[k];
        out[(size_t)kB * kT * kOut + rowbase + tid] = ssum;
    }
}

extern "C" void kernel_launch(void* const* d_in, const int* in_sizes, int n_in,
                              void* d_out, int out_size, void* d_ws, size_t ws_size,
                              hipStream_t stream) {
    (void)in_sizes; (void)n_in; (void)out_size; (void)d_ws; (void)ws_size;
    const float* x     = (const float*)d_in[0];
    const float* times = (const float*)d_in[1];
    const float* W_in  = (const float*)d_in[2];
    const float* b_in  = (const float*)d_in[3];
    const float* W1    = (const float*)d_in[4];
    const float* b1    = (const float*)d_in[5];
    const float* W2    = (const float*)d_in[6];
    const float* b2    = (const float*)d_in[7];
    const float* W_pk  = (const float*)d_in[8];
    const float* b_pk  = (const float*)d_in[9];
    const float* W_pd  = (const float*)d_in[10];
    const float* b_pd  = (const float*)d_in[11];
    float* out = (float*)d_out;

    hipFuncSetAttribute((const void*)node_rk4_kernel,
                        hipFuncAttributeMaxDynamicSharedMemorySize, SMEM_BYTES);
    node_rk4_kernel<<<kB / kRows, kThreads, SMEM_BYTES, stream>>>(
        x, times, W_in, b_in, W1, b1, W2, b2, W_pk, b_pk, W_pd, b_pd, out);
}
